// NDCG_Neighbor_Loss_55061480735166
// MI455X (gfx1250) — matrix operand order repaired and checked by reading, not compiled, not run on hardware
//
#include <hip/hip_runtime.h>
#include <hip/hip_bf16.h>
#include <math.h>

typedef __attribute__((ext_vector_type(16))) _Float16 v16h;
typedef __attribute__((ext_vector_type(8)))  _Float16 v8h;
typedef __attribute__((ext_vector_type(8)))  float    v8f;
typedef __attribute__((ext_vector_type(4)))  float    v4f;

#define B_SZ     64
#define M_ITEMS  128
#define NCAND    512
#define NPOS     64
#define U_COLS   129   // ITEM_NUM + 1
#define LN2F     0.69314718055994530942f
#define INV_LN2F 1.44269504088896340736f

// One workgroup per batch element b. 128 threads = 4 wave32; wave w owns
// j-rows [16w, 16w+16). Sequential scan over m=0..127 (dependency through u).
// Inner reduction g[j] = (1/512) * sum_c relu(p[c]-p[j]+1)^2 computed as a
// masked GEMM on v_wmma_f32_16x16x32_f16 using the expansion
//   sum relu^2 = sum m*q^2 - 2 p_j sum m*q + p_j^2 sum m,   q = p_c + 1,
// with each f32 feature split into exact f16 hi/lo halves (mask is exactly
// representable in f16, so products are exact; only f32 accumulation rounds).
// Prediction rows are double-buffered in LDS via global_load_async_to_lds_b128
// (ASYNCcnt), prefetching step m+1 while step m's WMMAs run.
__global__ __launch_bounds__(128)
void ndcg_scan_kernel(const float* __restrict__ preds,    // [64][128][512]
                      const float* __restrict__ rating,   // [64][128][64]
                      const float* __restrict__ npos_arr, // [64]
                      const float* __restrict__ idcg,     // [64][128]
                      const float* __restrict__ u_in,     // [500001][129]
                      const int*   __restrict__ user_id,  // [64]
                      const int*   __restrict__ item_id,  // [64][128][64]
                      float* __restrict__ s_out)          // [128][64]
{
    __shared__ __align__(16) float    p_buf[2][NCAND];    // double-buffered fp32 preds
    __shared__ __align__(16) _Float16 F_lds[6][NCAND];    // 0:(q^2)hi 1:(q^2)lo 2:q_hi 3:q_lo 4:ones 5:zeros
    __shared__ float Rtab[5][NPOS];                       // masked sums per j
    __shared__ float u_lds[U_COLS];                       // private u row for this b
    __shared__ float red[NPOS];

    const int t    = threadIdx.x;
    const int b    = blockIdx.x;
    const int lane = t & 31;
    const int wave = t >> 5;
    const int hi   = lane >> 4;       // half-wave select
    const int n    = lane & 15;       // N column (B/C/D) and A row within tile

    // ---- init private u row from global u (do not mutate the input) ----
    const long urow = (long)user_id[b] * U_COLS;
    for (int i = t; i < U_COLS; i += 128) u_lds[i] = u_in[urow + i];
    const float npos_b = npos_arr[b];

    // constant feature rows: 4 = ones (count column), 5 = zeros (unused columns)
    #pragma unroll
    for (int r = 0; r < 4; ++r) {
        const int c = t + 128 * r;
        F_lds[4][c] = (_Float16)1.0f;
        F_lds[5][c] = (_Float16)0.0f;
    }
    const int rowsel = (n < 4) ? n : ((n == 4) ? 4 : 5);
    const _Float16* __restrict__ Frow = &F_lds[rowsel][0];

    const float* pbase = preds + (long)b * M_ITEMS * NCAND;

    // ---- prologue: async-stage row m=0 into p_buf[0] (128 lanes x 16B = 2KB) ----
    {
        const unsigned ldst = (unsigned)(uintptr_t)&p_buf[0][t * 4];
        const float* gsrc = pbase + t * 4;
        asm volatile("global_load_async_to_lds_b128 %0, %1, off"
                     :: "v"(ldst), "v"(gsrc) : "memory");
    }

    int cur = 0;
    for (int m = 0; m < M_ITEMS; ++m) {
        // retire our async load from the previous step, then make it visible
        asm volatile("s_wait_asynccnt 0x0" ::: "memory");
        __syncthreads();
        const float* __restrict__ pc = &p_buf[cur][0];

        // ---- build hi/lo-split feature table from the staged row ----
        #pragma unroll
        for (int r = 0; r < 4; ++r) {
            const int c = t + 128 * r;
            const float p = pc[c];
            const float q = p + 1.0f;
            const float s = q * q;
            const _Float16 sh = (_Float16)s;
            const _Float16 qh = (_Float16)q;
            F_lds[0][c] = sh;
            F_lds[1][c] = (_Float16)(s - (float)sh);
            F_lds[2][c] = qh;
            F_lds[3][c] = (_Float16)(q - (float)qh);
        }

        // ---- kick off async staging of the next row while we compute ----
        if (m + 1 < M_ITEMS) {
            const unsigned ldst = (unsigned)(uintptr_t)&p_buf[cur ^ 1][t * 4];
            const float* gsrc = pbase + (long)(m + 1) * NCAND + t * 4;
            asm volatile("global_load_async_to_lds_b128 %0, %1, off"
                         :: "v"(ldst), "v"(gsrc) : "memory");
        }
        __syncthreads();

        // threshold for this wave's j-row mask: q_c > p_j  <=>  p_c > p_j - 1
        const float tj = pc[16 * wave + n] - 1.0f;

        // ---- masked GEMM: D(16x16) += mask(16x32) x F(32x16), K-chunks of 32 ----
        v8f acc = {0.f, 0.f, 0.f, 0.f, 0.f, 0.f, 0.f, 0.f};
        #pragma unroll
        for (int kc = 0; kc < 16; ++kc) {
            const int cb = kc * 32;

            // A (16x32 f16): lane lo holds K {0..7,16..23}, lane hi {8..15,24..31}
            const int ab = cb + (hi ? 8 : 0);
            const v4f* pp0 = reinterpret_cast<const v4f*>(&pc[ab]);
            const v4f* pp1 = reinterpret_cast<const v4f*>(&pc[ab + 16]);
            const v4f r0 = pp0[0], r1 = pp0[1];
            const v4f r2 = pp1[0], r3 = pp1[1];
            v16h a;
            #pragma unroll
            for (int i = 0; i < 4; ++i) {
                a[i]      = (_Float16)((r0[i] > tj) ? 1.0f : 0.0f);
                a[4 + i]  = (_Float16)((r1[i] > tj) ? 1.0f : 0.0f);
                a[8 + i]  = (_Float16)((r2[i] > tj) ? 1.0f : 0.0f);
                a[12 + i] = (_Float16)((r3[i] > tj) ? 1.0f : 0.0f);
            }

            // B (32x16 f16): lane lo holds K {0..7,16..23}, lane hi {8..15,24..31}; column n
            // (uniform control flow: constant columns come from LDS rows 4/5)
            const v8h* fp0 = reinterpret_cast<const v8h*>(&Frow[cb + 8 * hi]);
            const v8h* fp1 = reinterpret_cast<const v8h*>(&Frow[cb + 16 + 8 * hi]);
            const v8h f0 = fp0[0], f1 = fp1[0];
            v16h bv;
            #pragma unroll
            for (int i = 0; i < 8; ++i) { bv[i] = f0[i]; bv[8 + i] = f1[i]; }

            acc = __builtin_amdgcn_wmma_f32_16x16x32_f16(
                false, a, false, bv, (short)0, acc, false, false);
        }

        // ---- write the 5 useful D columns: row M = v + 8*hi, col = n ----
        if (n < 5) {
            #pragma unroll
            for (int v = 0; v < 8; ++v)
                Rtab[n][16 * wave + v + 8 * hi] = acc[v];
        }
        __syncthreads();

        // ---- per-j scoring: EMA update of u + nabla_f_g * g ----
        if (t < NPOS) {
            const float R0 = Rtab[0][t], R1 = Rtab[1][t];
            const float R2 = Rtab[2][t], R3 = Rtab[3][t];
            const float R4 = Rtab[4][t];
            const float pj = pc[t];
            const float gsum = (R0 + R1) - 2.0f * pj * (R2 + R3) + pj * pj * R4;
            const float g = gsum * (1.0f / 512.0f);

            const long base = ((long)b * M_ITEMS + m) * NPOS + t;
            const int iid = item_id[base];
            const float un = 0.9f * u_lds[iid] + 0.1f * g;   // EMA, gamma0=0.1
            u_lds[iid] = un;                                  // conflict-free scatter

            const float G  = exp2f(rating[base]) - 1.0f;
            const float x  = 128.0f * un;                     // M * g_u
            const float lt = log1pf(x) * INV_LN2F;
            const float nab = (G * 128.0f) / (lt * lt * (1.0f + x) * LN2F);
            red[t] = nab * g;
        }
        __syncthreads();

        if (t == 0) {
            float sum = 0.0f;
            #pragma unroll 8
            for (int j = 0; j < NPOS; ++j) sum += red[j];
            const float s = npos_b * (sum * (1.0f / 64.0f)) /
                            idcg[(long)b * M_ITEMS + m];
            s_out[m * B_SZ + b] = s;
        }
        cur ^= 1;
    }
}

// Second pass: tmp_m = mean_b s[m,b]; NaN-filtered mean over the 128 steps.
__global__ __launch_bounds__(128)
void ndcg_finalize_kernel(const float* __restrict__ s, float* __restrict__ out)
{
    __shared__ float ls[128];
    __shared__ float cs[128];
    const int m = threadIdx.x;
    float acc = 0.0f;
    #pragma unroll 8
    for (int b = 0; b < B_SZ; ++b) acc += s[m * B_SZ + b];
    const float tmp = acc * (1.0f / 64.0f);
    const bool nan = (tmp != tmp);
    ls[m] = nan ? 0.0f : tmp;
    cs[m] = nan ? 0.0f : 1.0f;
    __syncthreads();
    for (int off = 64; off > 0; off >>= 1) {
        if (m < off) { ls[m] += ls[m + off]; cs[m] += cs[m + off]; }
        __syncthreads();
    }
    if (m == 0) out[0] = ls[0] / cs[0];
}

extern "C" void kernel_launch(void* const* d_in, const int* in_sizes, int n_in,
                              void* d_out, int out_size, void* d_ws, size_t ws_size,
                              hipStream_t stream) {
    const float* preds   = (const float*)d_in[0];  // loc_predictions
    const float* rating  = (const float*)d_in[1];
    const float* nposarr = (const float*)d_in[2];  // num_pos_items
    const float* idcg    = (const float*)d_in[3];  // ideal_dcg
    const float* u_in    = (const float*)d_in[4];
    const int*   user_id = (const int*)d_in[5];
    const int*   item_id = (const int*)d_in[6];
    float* s_ws = (float*)d_ws;                    // 128*64 floats

    ndcg_scan_kernel<<<B_SZ, 128, 0, stream>>>(preds, rating, nposarr, idcg,
                                               u_in, user_id, item_id, s_ws);
    ndcg_finalize_kernel<<<1, 128, 0, stream>>>(s_ws, (float*)d_out);
}
